// MoEFFN_84705345011786
// MI455X (gfx1250) — matrix operand order repaired and checked by reading, not compiled, not run on hardware
//
#include <hip/hip_runtime.h>
#include <hip/hip_bf16.h>

#define D_MODEL 1024
#define D_FF    4096
#define NEXP    8
#define TOK     4096            // L*B = 512*8
#define CAP     640             // ceil(TOK/NEXP * 1.25)

#define BM 128
#define BN 256
#define BK 32
#define ASTR 56                 // A LDS row pitch in bf16 elems (112B: 16B-aligned, bank-spread)
#define BSTR 56                 // B(transposed) LDS row pitch in bf16 elems

typedef __bf16 bf16_t;
typedef bf16_t v16bf __attribute__((ext_vector_type(16)));
typedef float  v8f   __attribute__((ext_vector_type(8)));
typedef int    v4i   __attribute__((ext_vector_type(4)));
typedef __attribute__((address_space(1))) v4i* gv4i_p;   // global int4*
typedef __attribute__((address_space(3))) v4i* lv4i_p;   // LDS int4*

#if defined(__has_builtin)
#  if __has_builtin(__builtin_amdgcn_global_load_async_to_lds_b128)
#    define HAVE_ASYNC_B128 1
#  endif
#  if __has_builtin(__builtin_amdgcn_s_wait_asynccnt)
#    define HAVE_WAIT_ASYNC 1
#  endif
#endif

static __device__ __forceinline__ unsigned short f2bf(float f) {
  unsigned int u = __float_as_uint(f);
  u += 0x7FFFu + ((u >> 16) & 1u);      // round-to-nearest-even
  return (unsigned short)(u >> 16);
}

// ---------------- routing ----------------

__global__ __launch_bounds__(256) void router_kernel(
    const float* __restrict__ x, const float* __restrict__ rw,
    int* __restrict__ top1, float* __restrict__ prob)
{
  int wave = threadIdx.x >> 5;
  int lane = threadIdx.x & 31;
  int t = blockIdx.x * 8 + wave;
  if (t >= TOK) return;
  const float* xr = x + (size_t)t * D_MODEL;
  float acc[NEXP];
#pragma unroll
  for (int e = 0; e < NEXP; ++e) acc[e] = 0.f;
  for (int k = lane; k < D_MODEL; k += 32) {
    float xv = xr[k];
#pragma unroll
    for (int e = 0; e < NEXP; ++e)
      acc[e] = fmaf(xv, rw[e * D_MODEL + k], acc[e]);
  }
#pragma unroll
  for (int e = 0; e < NEXP; ++e) {
#pragma unroll
    for (int off = 16; off > 0; off >>= 1)
      acc[e] += __shfl_xor(acc[e], off, 32);
  }
  if (lane == 0) {
    int best = 0; float bl = acc[0];
#pragma unroll
    for (int e = 1; e < NEXP; ++e) if (acc[e] > bl) { bl = acc[e]; best = e; }
    float s = 0.f;
#pragma unroll
    for (int e = 0; e < NEXP; ++e) s += __expf(acc[e] - bl);
    top1[t] = best;
    prob[t] = 1.0f / s;                  // softmax value at the argmax
  }
}

__global__ void zero_kernel(int* counts, int* cursor) {
  if (threadIdx.x < NEXP) { counts[threadIdx.x] = 0; cursor[threadIdx.x] = 0; }
}

// rank of token within its expert group by (prob desc, index asc); capacity drop -> expert 0
__global__ __launch_bounds__(256) void rank_kernel(
    const int* __restrict__ top1, const float* __restrict__ prob,
    int* __restrict__ final_e, int* __restrict__ counts)
{
  __shared__ int   se[256];
  __shared__ float sp[256];
  int t = blockIdx.x * 256 + threadIdx.x;
  int   et = top1[t];
  float pt = prob[t];
  int rank = 0;
  for (int c = 0; c < TOK; c += 256) {
    se[threadIdx.x] = top1[c + threadIdx.x];
    sp[threadIdx.x] = prob[c + threadIdx.x];
    __syncthreads();
#pragma unroll 4
    for (int j = 0; j < 256; ++j) {
      int jg = c + j;
      bool ahead = (se[j] == et) && ((sp[j] > pt) || (sp[j] == pt && jg < t));
      rank += ahead ? 1 : 0;
    }
    __syncthreads();
  }
  int fe = (rank < CAP) ? et : 0;
  final_e[t] = fe;
  atomicAdd(&counts[fe], 1);
}

__global__ void prefix_kernel(const int* __restrict__ counts, int* __restrict__ starts) {
  if (threadIdx.x == 0) {
    int s = 0;
    for (int e = 0; e < NEXP; ++e) { starts[e] = s; s += counts[e]; }
    starts[NEXP] = s;
  }
}

__global__ __launch_bounds__(256) void scatter_kernel(
    const int* __restrict__ final_e, const int* __restrict__ starts,
    int* __restrict__ cursor, int* __restrict__ bucket)
{
  int t = blockIdx.x * 256 + threadIdx.x;
  int e = final_e[t];
  int slot = atomicAdd(&cursor[e], 1);
  bucket[starts[e] + slot] = t;
}

// ---------------- WMMA fragment loads (ISA 7.12.2 layouts) ----------------

// A 16x32 bf16: lanes 0-15 row M=lane, halves K[0..7]/K[16..23]; lanes 16-31: K[8..15]/K[24..31]
static __device__ __forceinline__ v16bf load_afrag(const unsigned short* As, int m, int lane) {
  int kq  = lane >> 4;
  const unsigned short* p = As + (m + (lane & 15)) * ASTR;
  union { v16bf v; uint4 q[2]; } u;
  u.q[0] = *(const uint4*)(p + kq * 8);
  u.q[1] = *(const uint4*)(p + 16 + kq * 8);
  return u.v;
}
// B 32x16 bf16 stored K-transposed in LDS (Bs[n][k]): lanes 0-15 hold K[0..7]/K[16..23], 16-31 K[8..15]/K[24..31]
static __device__ __forceinline__ v16bf load_bfrag(const unsigned short* Bs, int n, int lane) {
  int kq  = lane >> 4;
  const unsigned short* p = Bs + (n + (lane & 15)) * BSTR;
  union { v16bf v; uint4 q[2]; } u;
  u.q[0] = *(const uint4*)(p + kq * 8);
  u.q[1] = *(const uint4*)(p + 16 + kq * 8);
  return u.v;
}

// B tile loader: 32 x 256 f32 (row-major, row pitch ldn) -> bf16 K-transposed LDS.
// Loads two K rows per step and stores packed u32 (k,k+1) per n.
static __device__ __forceinline__ void load_btile_f32(
    unsigned short* Bs, const float* __restrict__ Wt, int k0, int n0, int ldn,
    int tid, int kmax)
{
  int n4    = (tid & 63) * 4;
  int kbase = (tid >> 6) * 2;          // 0,2,4,6
#pragma unroll
  for (int kk = kbase; kk < BK; kk += 8) {
    const float* r0 = Wt + (size_t)(k0 + kk) * ldn + n0 + n4;
    float4 f0 = *(const float4*)(r0);
    float4 f1 = *(const float4*)(r0 + ldn);
    unsigned int p0 = (unsigned)f2bf(f0.x) | ((unsigned)f2bf(f1.x) << 16);
    unsigned int p1 = (unsigned)f2bf(f0.y) | ((unsigned)f2bf(f1.y) << 16);
    unsigned int p2 = (unsigned)f2bf(f0.z) | ((unsigned)f2bf(f1.z) << 16);
    unsigned int p3 = (unsigned)f2bf(f0.w) | ((unsigned)f2bf(f1.w) << 16);
    *(unsigned int*)&Bs[(n4 + 0) * BSTR + kk] = p0;
    *(unsigned int*)&Bs[(n4 + 1) * BSTR + kk] = p1;
    *(unsigned int*)&Bs[(n4 + 2) * BSTR + kk] = p2;
    *(unsigned int*)&Bs[(n4 + 3) * BSTR + kk] = p3;
  }
  if (k0 + BK < kmax)  // hint: stream next weight chunk (global_prefetch_b8)
    __builtin_prefetch(Wt + (size_t)(k0 + BK + kbase) * ldn + n0 + n4, 0, 1);
}

// ---------------- GEMM 1: H = relu(gather(X) @ W1[e] + b1[e]) ----------------

__global__ __launch_bounds__(256) void gemm1_kernel(
    const float* __restrict__ x, const float* __restrict__ W1,
    const float* __restrict__ b1, unsigned short* __restrict__ H,
    const int* __restrict__ counts, const int* __restrict__ starts,
    const int* __restrict__ bucket)
{
  int e   = blockIdx.z;
  int cnt = counts[e];
  int m0  = blockIdx.y * BM;
  if (m0 >= cnt) return;
  int n0  = blockIdx.x * BN;
  int s0  = starts[e];

  __shared__ alignas(16) unsigned short As[BM * ASTR];
  __shared__ alignas(16) unsigned short Bs[BN * BSTR];
  __shared__ int rowtok[BM];

  int tid = threadIdx.x;
  if (tid < BM) {
    int r = m0 + tid;
    rowtok[tid] = (r < cnt) ? bucket[s0 + r] : bucket[s0];
  }
  __syncthreads();

  int lane = tid & 31;
  int wave = tid >> 5;
  int wm = (wave & 1) * 64;      // 2 waves along M
  int wn = (wave >> 1) * 64;     // 4 waves along N

  v8f zero = {0.f,0.f,0.f,0.f,0.f,0.f,0.f,0.f};
  v8f acc[4][4];
#pragma unroll
  for (int i = 0; i < 4; ++i)
#pragma unroll
    for (int j = 0; j < 4; ++j) acc[i][j] = zero;

  const float* W1e = W1 + (size_t)e * D_MODEL * D_FF;

  for (int k0 = 0; k0 < D_MODEL; k0 += BK) {
    { // A tile: gather 128x32 f32 -> bf16 LDS
      int r = tid >> 1, half = tid & 1;
      const float* src = x + (size_t)rowtok[r] * D_MODEL + k0 + half * 16;
      unsigned short* dst = As + r * ASTR + half * 16;
#pragma unroll
      for (int j = 0; j < 4; ++j) {
        float4 f = *(const float4*)(src + j * 4);
        uint2 p;
        p.x = (unsigned)f2bf(f.x) | ((unsigned)f2bf(f.y) << 16);
        p.y = (unsigned)f2bf(f.z) | ((unsigned)f2bf(f.w) << 16);
        *(uint2*)(dst + j * 4) = p;
      }
    }
    load_btile_f32(Bs, W1e, k0, n0, D_FF, tid, D_MODEL);
    __syncthreads();
    v16bf af[4], bfr[4];
#pragma unroll
    for (int i = 0; i < 4; ++i) af[i]  = load_afrag(As, wm + i * 16, lane);
#pragma unroll
    for (int j = 0; j < 4; ++j) bfr[j] = load_bfrag(Bs, wn + j * 16, lane);
#pragma unroll
    for (int i = 0; i < 4; ++i)
#pragma unroll
      for (int j = 0; j < 4; ++j)
        acc[i][j] = __builtin_amdgcn_wmma_f32_16x16x32_bf16(
            false, af[i], false, bfr[j], (short)0, acc[i][j], false, false);
    __syncthreads();
  }

  // epilogue: +bias, relu, bf16 store to H (compact expert-grouped rows)
  int col   = lane & 15;
  int rhalf = (lane >> 4) * 8;
#pragma unroll
  for (int j = 0; j < 4; ++j) {
    int n = n0 + wn + j * 16 + col;
    float bias = b1[e * D_FF + n];
#pragma unroll
    for (int i = 0; i < 4; ++i) {
#pragma unroll
      for (int r = 0; r < 8; ++r) {
        int mloc = wm + i * 16 + rhalf + r;
        if (m0 + mloc < cnt) {
          float v = acc[i][j][r] + bias;
          v = v > 0.f ? v : 0.f;
          H[(size_t)(s0 + m0 + mloc) * D_FF + n] = f2bf(v);
        }
      }
    }
  }
}

// ---------------- GEMM 2: out[tok] = H @ W2[e] + b2[e] ----------------

__global__ __launch_bounds__(256) void gemm2_kernel(
    const unsigned short* __restrict__ H, const float* __restrict__ W2,
    const float* __restrict__ b2, float* __restrict__ out,
    const int* __restrict__ counts, const int* __restrict__ starts,
    const int* __restrict__ bucket)
{
  int e   = blockIdx.z;
  int cnt = counts[e];
  int m0  = blockIdx.y * BM;
  if (m0 >= cnt) return;
  int n0  = blockIdx.x * BN;
  int s0  = starts[e];

  __shared__ alignas(16) unsigned short As[BM * ASTR];
  __shared__ alignas(16) unsigned short Bs[BN * BSTR];
  __shared__ int rowtok[BM];

  int tid = threadIdx.x;
  if (tid < BM) {
    int r = m0 + tid;
    rowtok[tid] = (r < cnt) ? bucket[s0 + r] : 0;
  }
  __syncthreads();

  int lane = tid & 31;
  int wave = tid >> 5;
  int wm = (wave & 1) * 64;
  int wn = (wave >> 1) * 64;

  v8f zero = {0.f,0.f,0.f,0.f,0.f,0.f,0.f,0.f};
  v8f acc[4][4];
#pragma unroll
  for (int i = 0; i < 4; ++i)
#pragma unroll
    for (int j = 0; j < 4; ++j) acc[i][j] = zero;

  const float* W2e = W2 + (size_t)e * D_FF * D_MODEL;

  for (int k0 = 0; k0 < D_FF; k0 += BK) {
    { // A tile: 128x32 bf16 copy H -> LDS (async DMA path on CDNA5)
      int r = tid >> 1, half = tid & 1;
      int rr = m0 + r;
      int srow = s0 + ((rr < cnt) ? rr : 0);
      const unsigned short* src = H + (size_t)srow * D_FF + k0 + half * 16;
      unsigned short* dst = As + r * ASTR + half * 16;
#ifdef HAVE_ASYNC_B128
      __builtin_amdgcn_global_load_async_to_lds_b128(
          (gv4i_p)(void*)(src), (lv4i_p)(void*)(dst), 0, 0);
      __builtin_amdgcn_global_load_async_to_lds_b128(
          (gv4i_p)(void*)(src + 8), (lv4i_p)(void*)(dst + 8), 0, 0);
#else
      *(uint4*)(dst)     = *(const uint4*)(src);
      *(uint4*)(dst + 8) = *(const uint4*)(src + 8);
#endif
    }
    load_btile_f32(Bs, W2e, k0, n0, D_MODEL, tid, D_FF);
#ifdef HAVE_ASYNC_B128
#  ifdef HAVE_WAIT_ASYNC
    __builtin_amdgcn_s_wait_asynccnt(0);
#  else
    asm volatile("s_wait_asynccnt 0x0" ::: "memory");
#  endif
#endif
    __syncthreads();
    v16bf af[4], bfr[4];
#pragma unroll
    for (int i = 0; i < 4; ++i) af[i]  = load_afrag(As, wm + i * 16, lane);
#pragma unroll
    for (int j = 0; j < 4; ++j) bfr[j] = load_bfrag(Bs, wn + j * 16, lane);
#pragma unroll
    for (int i = 0; i < 4; ++i)
#pragma unroll
      for (int j = 0; j < 4; ++j)
        acc[i][j] = __builtin_amdgcn_wmma_f32_16x16x32_bf16(
            false, af[i], false, bfr[j], (short)0, acc[i][j], false, false);
    __syncthreads();
  }

  // epilogue: +bias, scatter fp32 rows to each token's output row
  int col   = lane & 15;
  int rhalf = (lane >> 4) * 8;
#pragma unroll
  for (int j = 0; j < 4; ++j) {
    int n = n0 + wn + j * 16 + col;
    float bias = b2[e * D_MODEL + n];
#pragma unroll
    for (int i = 0; i < 4; ++i) {
#pragma unroll
      for (int r = 0; r < 8; ++r) {
        int mloc = wm + i * 16 + rhalf + r;
        if (m0 + mloc < cnt) {
          int tok = rowtok[mloc];
          out[(size_t)tok * D_MODEL + n] = acc[i][j][r] + bias;
        }
      }
    }
  }
}

// ---------------- launch ----------------

extern "C" void kernel_launch(void* const* d_in, const int* in_sizes, int n_in,
                              void* d_out, int out_size, void* d_ws, size_t ws_size,
                              hipStream_t stream)
{
  const float* x  = (const float*)d_in[0];
  const float* rw = (const float*)d_in[1];
  const float* W1 = (const float*)d_in[2];
  const float* b1 = (const float*)d_in[3];
  const float* W2 = (const float*)d_in[4];
  const float* b2 = (const float*)d_in[5];
  float* out = (float*)d_out;

  char* ws = (char*)d_ws;
  size_t off = 0;
  unsigned short* H = (unsigned short*)(ws + off); off += (size_t)TOK * D_FF * 2; // 32 MB bf16
  int*   top1    = (int*)(ws + off);   off += TOK * 4;
  float* prob    = (float*)(ws + off); off += TOK * 4;
  int*   final_e = (int*)(ws + off);   off += TOK * 4;
  int*   bucket  = (int*)(ws + off);   off += TOK * 4;
  int*   counts  = (int*)(ws + off);   off += NEXP * 4;
  int*   starts  = (int*)(ws + off);   off += (NEXP + 1) * 4;
  int*   cursor  = (int*)(ws + off);   off += NEXP * 4;

  zero_kernel  <<<1, 32, 0, stream>>>(counts, cursor);
  router_kernel<<<TOK / 8, 256, 0, stream>>>(x, rw, top1, prob);
  rank_kernel  <<<TOK / 256, 256, 0, stream>>>(top1, prob, final_e, counts);
  prefix_kernel<<<1, 32, 0, stream>>>(counts, starts);
  scatter_kernel<<<TOK / 256, 256, 0, stream>>>(final_e, starts, cursor, bucket);

  dim3 g1(D_FF / BN, TOK / BM, NEXP);
  gemm1_kernel<<<g1, 256, 0, stream>>>(x, W1, b1, H, counts, starts, bucket);
  dim3 g2(D_MODEL / BN, TOK / BM, NEXP);
  gemm2_kernel<<<g2, 256, 0, stream>>>(H, W2, b2, out, counts, starts, bucket);
}
